// CrossAttention_16441134809459
// MI455X (gfx1250) — matrix operand order repaired and checked by reading, not compiled, not run on hardware
//
#include <hip/hip_runtime.h>
#include <hip/hip_bf16.h>

// CDNA5 / gfx1250, wave32. Matmuls via V_WMMA_F32_16X16X32_BF16.
// bf16 operands staged to LDS with GLOBAL_LOAD_ASYNC_TO_LDS_B128 (ASYNCcnt),
// fragments loaded as contiguous 16B ds_load_b128 chunks (no per-use cvt).
typedef __attribute__((ext_vector_type(16))) __bf16 v16bf;
typedef __attribute__((ext_vector_type(8)))  __bf16 v8bf;
typedef __attribute__((ext_vector_type(4)))  __bf16 v4bf;
typedef __attribute__((ext_vector_type(2)))  __bf16 v2bf;
typedef __attribute__((ext_vector_type(8)))  float  v8f;

#define DEV __device__ __forceinline__

// LDS byte offset of a generic pointer to a __shared__ object (LDS aperture is
// 4GB-aligned on amdgcn, so low 32 bits are the LDS byte address).
DEV unsigned lds_off(const void* p) { return (unsigned)(uintptr_t)p; }

// Async 16B-per-lane copy: LDS[ldsoff] = MEM[gaddr].
DEV void async_b128(unsigned ldsoff, const void* gaddr) {
  asm volatile("global_load_async_to_lds_b128 %0, %1, off"
               :: "v"(ldsoff), "v"(gaddr) : "memory");
}

DEV v16bf concat8(v8bf lo, v8bf hi) {
  return __builtin_shufflevector(lo, hi, 0, 1, 2, 3, 4, 5, 6, 7,
                                         8, 9, 10, 11, 12, 13, 14, 15);
}

// ---------------------------------------------------------------------------
// bf16 fragment loaders (ISA 7.12.2 layouts). Both operands k-contiguous:
// A [.][lda] bf16 (MxK row-major): lane -> m=L%16, khalf=8*(L/16);
//   elements are chunks k=[kh,kh+7] and k=[kh+16,kh+23]  -> 2 x 16B loads.
// Bt [.][ldb] bf16 (NxK row-major = B^T): lane -> n=L%16, kg=8*(L/16);
//   elements are chunks k=[kg,kg+7] and k=[kg+16,kg+23]  -> 2 x 16B loads.
// C/D (16x16 f32): lane -> n=L%16; reg r -> m = r + 8*(L/16).
// ---------------------------------------------------------------------------
DEV v16bf frag_a_bf(const __bf16* __restrict__ A, int lda, int m0, int k0) {
  const int lane = threadIdx.x & 31;
  const int m  = lane & 15;
  const int kh = (lane >> 4) << 3;
  const __bf16* base = A + (long)(m0 + m) * lda + k0 + kh;
  v8bf lo = *(const v8bf*)(base);
  v8bf hi = *(const v8bf*)(base + 16);
  return concat8(lo, hi);
}

DEV v16bf frag_b_bf(const __bf16* __restrict__ Bt, int ldb, int n0, int k0) {
  const int lane = threadIdx.x & 31;
  const int n  = lane & 15;
  const int kg = (lane >> 4) << 3;
  const __bf16* base = Bt + (long)(n0 + n) * ldb + k0 + kg;
  v8bf lo = *(const v8bf*)(base);
  v8bf hi = *(const v8bf*)(base + 16);
  return concat8(lo, hi);
}

// f32 A-fragment loader (used only for the P restage buffer in attention).
DEV v16bf frag_a_f32(const float* __restrict__ A, int lda, int m0, int k0) {
  const int lane = threadIdx.x & 31;
  const int m  = lane & 15;
  const int kh = (lane >> 4) << 3;
  const float* base = A + (long)(m0 + m) * lda + k0 + kh;
  v16bf f;
#pragma unroll
  for (int r = 0; r < 8; ++r) {
    const int kr = ((r >> 2) << 4) + ((r & 3) << 1);
    f[2 * r]     = (__bf16)base[kr];
    f[2 * r + 1] = (__bf16)base[kr + 1];
  }
  return f;
}

DEV v8f wmma_bf16(v16bf a, v16bf b, v8f c) {
  return __builtin_amdgcn_wmma_f32_16x16x32_bf16(false, a, false, b,
                                                 (short)0, c, false, false);
}

// ---------------------------------------------------------------------------
// Pre-pass 1: flat f32 -> bf16 (weights). nelem % 4 == 0.
// ---------------------------------------------------------------------------
__global__ void __launch_bounds__(256)
cvt_bf16_flat(const float* __restrict__ in, __bf16* __restrict__ out, long nelem) {
  const long i = ((long)blockIdx.x * 256 + threadIdx.x) * 4;
  if (i >= nelem) return;
  const float4 v = *(const float4*)(in + i);
  v4bf o; o[0] = (__bf16)v.x; o[1] = (__bf16)v.y; o[2] = (__bf16)v.z; o[3] = (__bf16)v.w;
  *(v4bf*)(out + i) = o;
}

// ---------------------------------------------------------------------------
// Pre-pass 2: [C][S] f32 -> [S][C] bf16 transpose-convert (per batch z).
// grid (S/32, C/32, batch), block (32, 8).
// ---------------------------------------------------------------------------
__global__ void __launch_bounds__(256)
transpose_cvt(const float* __restrict__ in, __bf16* __restrict__ out,
              int C, int S) {
  __shared__ float tile[32][33];
  const int s0 = blockIdx.x * 32, c0 = blockIdx.y * 32;
  const long bz = blockIdx.z;
  const float* ib = in + bz * (long)C * S;
  __bf16* ob = out + bz * (long)S * C;
  const int tx = threadIdx.x, ty = threadIdx.y;
#pragma unroll
  for (int i = 0; i < 32; i += 8)
    tile[ty + i][tx] = ib[(long)(c0 + ty + i) * S + s0 + tx];
  __syncthreads();
#pragma unroll
  for (int i = 0; i < 32; i += 8)
    ob[(long)(s0 + ty + i) * C + c0 + tx] = (__bf16)tile[tx][ty + i];
}

// ---------------------------------------------------------------------------
// GEMM: C[m,n] = alpha * sum_k A[m,k] * B[k,n]
//   A  : [M][K] bf16 (lda = K), Bt : [N][K] bf16 (ldb = K)  -- k-contiguous
//   out: bf16 at Cbf[m*so_m + n*so_n]  (so_m==1 -> packed pair stores)
//        or f32 at Cf32[m*so_m + n*so_n]
// 128 threads = 4 waves, 64x64 tile, BK=32, double-buffered async staging.
// All fragments are loaded before the wmma group -> one dscnt drain, 4 wmma
// back-to-back.
// ---------------------------------------------------------------------------
__global__ void __launch_bounds__(128)
gemm_bf16_wmma(const __bf16* __restrict__ A, const __bf16* __restrict__ Bt,
               __bf16* __restrict__ Cbf, float* __restrict__ Cf32,
               int K, long b_batch, long c_batch,
               long so_m, long so_n, float alpha) {
  __shared__ __align__(16) __bf16 sA[2][64 * 32];  // [m][k] 4KB each
  __shared__ __align__(16) __bf16 sB[2][64 * 32];  // [n][k] 4KB each
  const int tid  = threadIdx.x;
  const int wave = tid >> 5;
  const int lane = tid & 31;
  const int n0   = blockIdx.x * 64;
  const int m0   = blockIdx.y * 64;
  const long bz  = blockIdx.z;
  const __bf16* Bb = Bt + bz * b_batch;
  const int nk = K >> 5;

  auto issue = [&](int buf, int k0) {
    // 64 rows x 32 bf16 = 64B/row -> 4 granules/row, 256 granules, 2/thread.
#pragma unroll
    for (int i = 0; i < 2; ++i) {
      const int g = tid + i * 128;
      const int row = g >> 2, col = (g & 3) << 3;
      async_b128(lds_off(&sA[buf][(row << 5) + col]),
                 A + (long)(m0 + row) * K + k0 + col);
      async_b128(lds_off(&sB[buf][(row << 5) + col]),
                 Bb + (long)(n0 + row) * K + k0 + col);
    }
  };

  v8f acc[4];
#pragma unroll
  for (int t = 0; t < 4; ++t) acc[t] = (v8f){};

  auto compute = [&](int cur) {
    const v16bf fa  = frag_a_bf(&sA[cur][0], 32, wave << 4, 0);
    v16bf fb[4];
#pragma unroll
    for (int t = 0; t < 4; ++t)
      fb[t] = frag_b_bf(&sB[cur][0], 32, t << 4, 0);
#pragma unroll
    for (int t = 0; t < 4; ++t)
      acc[t] = wmma_bf16(fa, fb[t], acc[t]);
  };

  // Steady-state loop has uniform control flow; last iteration peeled.
  issue(0, 0);
  for (int kt = 0; kt < nk - 1; ++kt) {
    const int cur = kt & 1;
    issue(cur ^ 1, (kt + 1) << 5);
    asm volatile("s_wait_asynccnt 4" ::: "memory");  // prev buffer landed
    __syncthreads();
    compute(cur);
    __syncthreads();
  }
  asm volatile("s_wait_asynccnt 0" ::: "memory");
  __syncthreads();
  compute((nk - 1) & 1);

  const int nn = lane & 15;
  const int mh = (lane >> 4) << 3;
  const int mrow0 = m0 + (wave << 4) + mh;
  if (Cf32) {
    float* cb = Cf32 + bz * c_batch;
#pragma unroll
    for (int t = 0; t < 4; ++t)
#pragma unroll
      for (int r = 0; r < 8; ++r)
        cb[(long)(mrow0 + r) * so_m + (long)(n0 + (t << 4) + nn) * so_n] =
            alpha * acc[t][r];
  } else if (so_m == 1) {
    // m-contiguous bf16 layout: pack pairs along r.
    __bf16* cb = Cbf + bz * c_batch;
#pragma unroll
    for (int t = 0; t < 4; ++t) {
      __bf16* p = cb + (long)(n0 + (t << 4) + nn) * so_n + mrow0;
#pragma unroll
      for (int r2 = 0; r2 < 4; ++r2) {
        v2bf two;
        two[0] = (__bf16)(alpha * acc[t][2 * r2]);
        two[1] = (__bf16)(alpha * acc[t][2 * r2 + 1]);
        *(v2bf*)(p + 2 * r2) = two;
      }
    }
  } else {
    __bf16* cb = Cbf + bz * c_batch;
#pragma unroll
    for (int t = 0; t < 4; ++t)
#pragma unroll
      for (int r = 0; r < 8; ++r)
        cb[(long)(mrow0 + r) * so_m + (long)(n0 + (t << 4) + nn) * so_n] =
            (__bf16)(alpha * acc[t][r]);
  }
}

// ---------------------------------------------------------------------------
// Flash attention, n=1024, d=64/head. Block = 4 waves sharing async-staged
// bf16 K/V tiles (double-buffered); each wave owns one 16-row i tile.
//   Qt : [b][1024 i][512 hd] bf16 (pre-scaled by 1/8)
//   Kt : [b][1024 j][512 hd] bf16
//   Vt : [b][512 hd][1024 j] bf16
//   ATTt:[b][1024 i][512 hd] bf16   ('b (x y) (h d)' -> B^T for out-proj)
// ---------------------------------------------------------------------------
__global__ void __launch_bounds__(128)
attn_flash(const __bf16* __restrict__ Qt, const __bf16* __restrict__ Kt,
           const __bf16* __restrict__ Vt, __bf16* __restrict__ ATTt) {
  __shared__ __align__(16) __bf16 sK[2][32 * 64];   // [j][d] 4KB each
  __shared__ __align__(16) __bf16 sV[2][64 * 32];   // [d][j] 4KB each
  __shared__ __align__(16) float  pbuf[4][16 * 32]; // per-wave P restage
  const int tid  = threadIdx.x;
  const int wave = tid >> 5;
  const int lane = tid & 31;
  const int b    = blockIdx.z;
  const int h    = blockIdx.y;
  const int i0   = (blockIdx.x * 4 + wave) << 4;

  const __bf16* qb = Qt + (long)b * 1024 * 512 + h * 64;               // [i][d]+
  const __bf16* kb = Kt + (long)b * 1024 * 512 + h * 64;               // [j][d]+
  const __bf16* vb = Vt + (long)b * 512 * 1024 + (long)h * 64 * 1024;  // [d][j]
  __bf16*      atb = ATTt + (long)b * 1024 * 512 + h * 64;

  auto issue = [&](int buf, int j0) {
    // K: 32 rows x 64 bf16 = 128B/row -> 8 granules/row, 256 total, 2/thread.
#pragma unroll
    for (int i = 0; i < 2; ++i) {
      const int g = tid + i * 128;
      const int row = g >> 3, col = (g & 7) << 3;
      async_b128(lds_off(&sK[buf][(row << 6) + col]),
                 kb + (long)(j0 + row) * 512 + col);
    }
    // V: 64 rows x 32 bf16 = 64B/row -> 4 granules/row, 256 total, 2/thread.
#pragma unroll
    for (int i = 0; i < 2; ++i) {
      const int g = tid + i * 128;
      const int row = g >> 2, col = (g & 3) << 3;
      async_b128(lds_off(&sV[buf][(row << 5) + col]),
                 vb + (long)row * 1024 + j0 + col);
    }
  };

  float mrow[8], lrow[8];
#pragma unroll
  for (int r = 0; r < 8; ++r) { mrow[r] = -3.0e38f; lrow[r] = 0.0f; }
  v8f acc[4];
#pragma unroll
  for (int t = 0; t < 4; ++t) acc[t] = (v8f){};

  // Q fragments: A[m=i, k=d], lda = 512. K=64 -> 2 frags (global 16B loads).
  const v16bf qa0 = frag_a_bf(qb, 512, i0, 0);
  const v16bf qa1 = frag_a_bf(qb, 512, i0, 32);

  auto step = [&](int cur) {
    // ---- S = Q K^T, 32 columns; Bt = sK[j][d] (ldb = 64) ----
    // Load all 4 K fragments first -> single dscnt drain, 4 wmma in a row.
    const __bf16* Ktl = &sK[cur][0];
    v16bf kf[4];
    kf[0] = frag_b_bf(Ktl, 64, 0,  0);
    kf[1] = frag_b_bf(Ktl, 64, 16, 0);
    kf[2] = frag_b_bf(Ktl, 64, 0,  32);
    kf[3] = frag_b_bf(Ktl, 64, 16, 32);
    v8f s0 = {}, s1 = {};
    s0 = wmma_bf16(qa0, kf[0], s0);
    s1 = wmma_bf16(qa0, kf[1], s1);
    s0 = wmma_bf16(qa1, kf[2], s0);
    s1 = wmma_bf16(qa1, kf[3], s1);

    // ---- online softmax (row = r + 8*(lane/16); cols across 16-lane group) ----
    float scale[8];
#pragma unroll
    for (int r = 0; r < 8; ++r) {
      float t = fmaxf(s0[r], s1[r]);
#pragma unroll
      for (int off = 1; off < 16; off <<= 1)
        t = fmaxf(t, __shfl_xor(t, off, 16));
      const float mn = fmaxf(mrow[r], t);
      scale[r] = __expf(mrow[r] - mn);
      mrow[r]  = mn;
      const float p0 = __expf(s0[r] - mn);
      const float p1 = __expf(s1[r] - mn);
      s0[r] = p0; s1[r] = p1;
      float rs = p0 + p1;
#pragma unroll
      for (int off = 1; off < 16; off <<= 1)
        rs += __shfl_xor(rs, off, 16);
      lrow[r] = lrow[r] * scale[r] + rs;
    }
#pragma unroll
    for (int t = 0; t < 4; ++t)
#pragma unroll
      for (int r = 0; r < 8; ++r)
        acc[t][r] *= scale[r];

    // ---- restage P (16x32) D-layout -> LDS -> A-layout ----
    {
      const int nn = lane & 15;
      const int mh = (lane >> 4) << 3;
      float* pb = &pbuf[wave][0];
#pragma unroll
      for (int r = 0; r < 8; ++r) {
        pb[(mh + r) * 32 + nn]      = s0[r];
        pb[(mh + r) * 32 + 16 + nn] = s1[r];
      }
    }
    asm volatile("s_wait_dscnt 0" ::: "memory");   // same-wave LDS RAW fence
    const v16bf pa = frag_a_f32(&pbuf[wave][0], 32, 0, 0);

    // ---- acc += P @ V ; Bt = sV[d][j] (ldb = 32) ----
    v16bf vf[4];
#pragma unroll
    for (int t = 0; t < 4; ++t)
      vf[t] = frag_b_bf(&sV[cur][0], 32, t << 4, 0);
#pragma unroll
    for (int t = 0; t < 4; ++t)
      acc[t] = wmma_bf16(pa, vf[t], acc[t]);
  };

  issue(0, 0);
  for (int jt = 0; jt < 31; ++jt) {
    const int cur = jt & 1;
    issue(cur ^ 1, (jt + 1) << 5);
    asm volatile("s_wait_asynccnt 4" ::: "memory");
    __syncthreads();
    step(cur);
    __syncthreads();
  }
  asm volatile("s_wait_asynccnt 0" ::: "memory");
  __syncthreads();
  step(1);  // jt = 31 -> cur = 1

  // ---- epilogue: ATTt[i][h*64+d] = acc / l ----
  const int nn = lane & 15;
  const int mh = (lane >> 4) << 3;
  float inv[8];
#pragma unroll
  for (int r = 0; r < 8; ++r) inv[r] = 1.0f / lrow[r];
#pragma unroll
  for (int t = 0; t < 4; ++t)
#pragma unroll
    for (int r = 0; r < 8; ++r)
      atb[(long)(i0 + mh + r) * 512 + (t << 4) + nn] =
          (__bf16)(acc[t][r] * inv[r]);
}

// ---------------------------------------------------------------------------
// Launch. Workspace (bf16): wq | wkv | wout | xqT | xkvT | Qt | Kt | Vt | ATTt
// = 0.31 + 1.25 + 0.31 + 5 + 10 + 8 + 8 + 8 + 8 MiB  ~= 49 MiB.
// ---------------------------------------------------------------------------
extern "C" void kernel_launch(void* const* d_in, const int* in_sizes, int n_in,
                              void* d_out, int out_size, void* d_ws, size_t ws_size,
                              hipStream_t stream) {
  const float* x_q   = (const float*)d_in[0];   // [8, 320, 1024]
  const float* x_kv  = (const float*)d_in[1];   // [8, 640, 1024]
  const float* w_q   = (const float*)d_in[2];   // [512, 320]
  const float* w_kv  = (const float*)d_in[3];   // [1024, 640]
  const float* w_out = (const float*)d_in[4];   // [320, 512]
  float* out = (float*)d_out;                   // [8, 320, 1024]

  __bf16* p = (__bf16*)d_ws;
  __bf16* wq_bf   = p; p += (size_t)512 * 320;
  __bf16* wkv_bf  = p; p += (size_t)1024 * 640;
  __bf16* wout_bf = p; p += (size_t)320 * 512;
  __bf16* xqT     = p; p += (size_t)8 * 1024 * 320;   // [b][s][c]
  __bf16* xkvT    = p; p += (size_t)8 * 1024 * 640;   // [b][s][c]
  __bf16* Qt      = p; p += (size_t)8 * 1024 * 512;   // [b][i][hd]
  __bf16* Kt      = p; p += (size_t)8 * 1024 * 512;   // [b][j][hd]
  __bf16* Vt      = p; p += (size_t)8 * 512 * 1024;   // [b][hd][j]
  __bf16* ATTt    = p;                                 // [b][i][hd]

  // -- pre-pass: weights -> bf16; activations -> transposed bf16 --
  cvt_bf16_flat<<<dim3((512 * 320) / 1024), 256, 0, stream>>>(w_q, wq_bf, (long)512 * 320);
  cvt_bf16_flat<<<dim3((1024 * 640) / 1024), 256, 0, stream>>>(w_kv, wkv_bf, (long)1024 * 640);
  cvt_bf16_flat<<<dim3((320 * 512) / 1024), 256, 0, stream>>>(w_out, wout_bf, (long)320 * 512);
  transpose_cvt<<<dim3(32, 10, 8), dim3(32, 8), 0, stream>>>(x_q, xqT, 320, 1024);
  transpose_cvt<<<dim3(32, 20, 8), dim3(32, 8), 0, stream>>>(x_kv, xkvT, 640, 1024);

  const dim3 blk(128, 1, 1);

  // Qt[b][s][o] = (1/8) * (w_q @ x_q[b])^T    M=512 N=1024 K=320
  gemm_bf16_wmma<<<dim3(16, 8, 8), blk, 0, stream>>>(
      wq_bf, xqT, Qt, nullptr, 320,
      (long)1024 * 320, (long)1024 * 512, /*so_m=*/1, /*so_n=*/512, 0.125f);

  // Kt[b][s][o] = (w_kv[0:512] @ x_kv[b])^T   M=512 N=1024 K=640
  gemm_bf16_wmma<<<dim3(16, 8, 8), blk, 0, stream>>>(
      wkv_bf, xkvT, Kt, nullptr, 640,
      (long)1024 * 640, (long)1024 * 512, /*so_m=*/1, /*so_n=*/512, 1.0f);

  // Vt[b][o][s] = w_kv[512:1024] @ x_kv[b]    M=512 N=1024 K=640
  gemm_bf16_wmma<<<dim3(16, 8, 8), blk, 0, stream>>>(
      wkv_bf + (size_t)512 * 640, xkvT, Vt, nullptr, 640,
      (long)1024 * 640, (long)512 * 1024, /*so_m=*/1024, /*so_n=*/1, 1.0f);

  // attention: ATTt[b][i][hd]
  attn_flash<<<dim3(16, 8, 8), blk, 0, stream>>>(Qt, Kt, Vt, ATTt);

  // out[b][o][s] = w_out @ ATTt[b]^T           M=320 N=1024 K=512, f32 out
  gemm_bf16_wmma<<<dim3(16, 5, 8), blk, 0, stream>>>(
      wout_bf, ATTt, nullptr, out, 512,
      (long)1024 * 512, (long)320 * 1024, /*so_m=*/1024, /*so_n=*/1, 1.0f);
}
